// _GraphConvNet_71571335021250
// MI455X (gfx1250) — matrix operand order repaired and checked by reading, not compiled, not run on hardware
//
#include <hip/hip_runtime.h>

#define HID 128
#define NG  64
#define LDP (HID + 8)   // padded LDS row in halves: 272B = 16B-aligned, bank-spread

typedef __attribute__((ext_vector_type(16))) _Float16 v16h;
typedef __attribute__((ext_vector_type(8)))  _Float16 v8h;
typedef __attribute__((ext_vector_type(4)))  _Float16 v4h;
typedef __attribute__((ext_vector_type(8)))  float    v8f;

// ---------------- elementwise helpers (float4-wide) ----------------
__global__ void __launch_bounds__(256) k_fill4(float4* __restrict__ p, int n4, float v) {
  int i = blockIdx.x * 256 + threadIdx.x;
  if (i < n4) p[i] = make_float4(v, v, v, v);
}

__global__ void __launch_bounds__(256) k_deg(const int* __restrict__ dst,
                                             float* __restrict__ deg, int ne) {
  int i = blockIdx.x * 256 + threadIdx.x;
  if (i < ne) atomicAdd(deg + dst[i], 1.0f);
}

__global__ void __launch_bounds__(256) k_rsqrt(float* __restrict__ p, int n) {
  int i = blockIdx.x * 256 + threadIdx.x;
  if (i < n) p[i] = rsqrtf(p[i]);   // deg >= 1 (self loop) so always valid
}

// ---------------- WMMA GEMM: Y[16*n_tiles x 128] = X @ W[128 x 128] ----------------
// 256 threads = 8 waves; W staged in LDS once per block (f16, transposed);
// grid-stride loop over 16-row tiles; wave w owns output cols [16w, 16w+16).
// Callers guarantee row count is a multiple of 16.
__global__ void __launch_bounds__(256) k_gemm128(const float* __restrict__ X,
                                                 const float* __restrict__ W,
                                                 float* __restrict__ Y, int n_tiles) {
  __shared__ __align__(16) _Float16 sA[16][LDP];    // row-major tile  [m][k]
  __shared__ __align__(16) _Float16 sWt[HID][LDP];  // transposed W    [n][k]
  const int tid = threadIdx.x;

  for (int i = tid; i < HID * HID; i += 256) {      // W is row-major [k][n]
    int k = i >> 7, n = i & (HID - 1);
    sWt[n][k] = (_Float16)W[i];
  }

  const int lane = tid & 31;
  const int col0 = (tid >> 5) * 16;
  const int m    = lane & 15;    // A row / B col within tile
  const int kg   = lane >> 4;    // lane-group per ISA 16-bit layouts

  for (int t = blockIdx.x; t < n_tiles; t += gridDim.x) {
    __syncthreads();                                // sWt ready / sA free
    const size_t rbase = (size_t)t * 16 * HID;
    const float4* Xv = (const float4*)(X + rbase);  // coalesced b128 staging
    #pragma unroll
    for (int i = tid; i < 16 * HID / 4; i += 256) { // 512 float4 -> 2 iters
      float4 xv = Xv[i];
      v4h h; h[0] = (_Float16)xv.x; h[1] = (_Float16)xv.y;
             h[2] = (_Float16)xv.z; h[3] = (_Float16)xv.w;
      int r = i >> 5, k4 = (i & 31) << 2;
      *(v4h*)&sA[r][k4] = h;                        // b64 store, bank-spread
    }
    __syncthreads();

    // prefetch next tile's X rows while this tile computes (global_prefetch_b8)
    if (t + (int)gridDim.x < n_tiles) {
      const char* nxt = (const char*)(X + rbase + (size_t)gridDim.x * 16 * HID);
      __builtin_prefetch(nxt + tid * 32, 0, 1);
    }

    v8f c = {};
    #pragma unroll
    for (int k0 = 0; k0 < HID; k0 += 32) {
      union { v16h v; v8h h[2]; } A, B;
      // A 16x32 f16: lane m holds K = k0 + kg*16 + 0..15 (contiguous, matches B)
      A.h[0] = *(const v8h*)&sA[m][k0 + kg * 16];
      A.h[1] = *(const v8h*)&sA[m][k0 + kg * 16 + 8];
      // B 32x16 f16: lane n holds K = k0 + kg*16 + 0..15 (contiguous in sWt)
      B.h[0] = *(const v8h*)&sWt[col0 + m][k0 + kg * 16];
      B.h[1] = *(const v8h*)&sWt[col0 + m][k0 + kg * 16 + 8];
      c = __builtin_amdgcn_wmma_f32_16x16x32_f16(false, A.v, false, B.v,
                                                 (short)0, c, false, false);
    }
    // D layout: M = kg*8 + r, N = m; rows always in range (n_rows % 16 == 0)
    float* yb = Y + rbase + (size_t)(kg * 8) * HID + col0 + m;
    #pragma unroll
    for (int r = 0; r < 8; ++r) yb[(size_t)r * HID] = c[r];
  }
}

// ---------------- edge scatter: agg[dst] += H[src] * dinv[src]*dinv[dst] ----------------
__global__ void __launch_bounds__(256) k_scatter(const float* __restrict__ H,
                                                 const int* __restrict__ src,
                                                 const int* __restrict__ dst,
                                                 const float* __restrict__ dinv,
                                                 float* __restrict__ agg, int ne) {
  int e    = (blockIdx.x * 256 + threadIdx.x) >> 5;   // one wave32 per edge
  int lane = threadIdx.x & 31;
  if (e >= ne) return;
  int s = src[e], d = dst[e];
  float nrm = dinv[s] * dinv[d];
  float4 v  = ((const float4*)(H + (size_t)s * HID))[lane];   // 128 = 32 lanes x 4
  float* o  = agg + (size_t)d * HID + lane * 4;
  atomicAdd(o + 0, v.x * nrm);
  atomicAdd(o + 1, v.y * nrm);
  atomicAdd(o + 2, v.z * nrm);
  atomicAdd(o + 3, v.w * nrm);
}

// out = relu(agg + selfH * dinv^2 + bias)   (fuses self-loop edge + bias + ReLU)
__global__ void __launch_bounds__(256) k_finalize4(const float4* __restrict__ agg,
                                                   const float4* __restrict__ selfH,
                                                   const float* __restrict__ dinv,
                                                   const float4* __restrict__ bias4,
                                                   float4* __restrict__ out, int n4) {
  int idx = blockIdx.x * 256 + threadIdx.x;
  if (idx >= n4) return;
  int i = idx >> 5, c4 = idx & 31;        // 32 float4 per 128-wide row
  float di = dinv[i], d2 = di * di;
  float4 a = agg[idx], s = selfH[idx], b = bias4[c4];
  float4 r;
  r.x = fmaxf(a.x + s.x * d2 + b.x, 0.0f);
  r.y = fmaxf(a.y + s.y * d2 + b.y, 0.0f);
  r.z = fmaxf(a.z + s.z * d2 + b.z, 0.0f);
  r.w = fmaxf(a.w + s.w * d2 + b.w, 0.0f);
  out[idx] = r;
}

// ---------------- pooling ----------------
__global__ void __launch_bounds__(256) k_pool_sum4(const float4* __restrict__ h,
                                                   const int* __restrict__ batch,
                                                   float* __restrict__ pooled, int n4) {
  int idx = blockIdx.x * 256 + threadIdx.x;
  if (idx >= n4) return;
  int i = idx >> 5, c4 = (idx & 31) << 2;
  float4 v = h[idx];
  float* base = pooled + batch[i] * HID + c4;
  atomicAdd(base + 0, v.x);
  atomicAdd(base + 1, v.y);
  atomicAdd(base + 2, v.z);
  atomicAdd(base + 3, v.w);
}

__global__ void __launch_bounds__(256) k_pool_cnt(const int* __restrict__ batch,
                                                  float* __restrict__ cnt, int n) {
  int i = blockIdx.x * 256 + threadIdx.x;
  if (i < n) atomicAdd(cnt + batch[i], 1.0f);
}

__global__ void __launch_bounds__(256) k_pool_div4(float4* __restrict__ pooled,
                                                   const float* __restrict__ cnt, int n4) {
  int idx = blockIdx.x * 256 + threadIdx.x;
  if (idx >= n4) return;
  float c = fmaxf(cnt[idx >> 5], 1.0f);
  float4 p = pooled[idx];
  p.x /= c; p.y /= c; p.z /= c; p.w /= c;
  pooled[idx] = p;
}

// ---------------- MLP head ----------------
__global__ void __launch_bounds__(256) k_bias_relu4(const float4* __restrict__ in,
                                                    const float4* __restrict__ bias4,
                                                    float4* __restrict__ out, int n4) {
  int idx = blockIdx.x * 256 + threadIdx.x;
  if (idx >= n4) return;
  float4 a = in[idx], b = bias4[idx & 31];
  float4 r;
  r.x = fmaxf(a.x + b.x, 0.0f);
  r.y = fmaxf(a.y + b.y, 0.0f);
  r.z = fmaxf(a.z + b.z, 0.0f);
  r.w = fmaxf(a.w + b.w, 0.0f);
  out[idx] = r;
}

__global__ void k_head(const float* __restrict__ z, const float* __restrict__ Wh2,
                       const float* __restrict__ bh2, float* __restrict__ out, int g) {
  int i = blockIdx.x * blockDim.x + threadIdx.x;
  if (i >= g) return;
  float s = bh2[0];
  #pragma unroll 8
  for (int k = 0; k < HID; ++k) s += z[i * HID + k] * Wh2[k];
  out[i] = 1.0f / (1.0f + __expf(-s));
}

// ---------------- driver ----------------
extern "C" void kernel_launch(void* const* d_in, const int* in_sizes, int n_in,
                              void* d_out, int out_size, void* d_ws, size_t ws_size,
                              hipStream_t stream) {
  (void)n_in; (void)out_size; (void)ws_size;
  const float* x   = (const float*)d_in[0];
  const int*   ei  = (const int*)d_in[1];
  const int*   bat = (const int*)d_in[2];
  const float* W1  = (const float*)d_in[3];
  const float* b1  = (const float*)d_in[4];
  const float* W2  = (const float*)d_in[5];
  const float* b2  = (const float*)d_in[6];
  const float* Wh1 = (const float*)d_in[7];
  const float* bh1 = (const float*)d_in[8];
  const float* Wh2 = (const float*)d_in[9];
  const float* bh2 = (const float*)d_in[10];
  float* out = (float*)d_out;

  const int N = in_sizes[0] / HID;   // 50000, multiple of 16
  const int E = in_sizes[1] / 2;
  const int* src = ei;       // edge_index[0]
  const int* dst = ei + E;   // edge_index[1]

  float* ws = (float*)d_ws;
  const size_t NH = (size_t)N * HID;
  float* A      = ws;                  // N*HID  ping      (16B-aligned offsets)
  float* B      = A + NH;              // N*HID  pong
  float* dinv   = B + NH;              // N
  float* pooled = dinv + N;            // NG*HID   } contiguous with cnt,
  float* cnt    = pooled + NG * HID;   // NG       } zeroed together
  float* ztmp   = cnt + NG;            // NG*HID
  float* z      = ztmp + NG * HID;     // NG*HID

  auto cb = [](long n) { return (int)((n + 255) / 256); };

  const int n_tiles = N / 16;                        // 3125
  const int ggrid   = n_tiles < 640 ? n_tiles : 640; // W staged once per block
  const int NH4     = (int)(NH / 4);
  const int GH4     = NG * HID / 4;

  // degrees (incl. self loop) -> dinv = deg^{-1/2}
  k_fill4 <<<cb(N / 4 + 1), 256, 0, stream>>>((float4*)dinv, N / 4, 1.0f);
  k_deg   <<<cb(E), 256, 0, stream>>>(dst, dinv, E);
  k_rsqrt <<<cb(N), 256, 0, stream>>>(dinv, N);

  // ----- GCN layer 1 -----
  k_gemm128  <<<ggrid, 256, 0, stream>>>(x, W1, A, n_tiles);
  k_fill4    <<<cb(NH4), 256, 0, stream>>>((float4*)B, NH4, 0.0f);
  k_scatter  <<<cb((long)E * 32), 256, 0, stream>>>(A, src, dst, dinv, B, E);
  k_finalize4<<<cb(NH4), 256, 0, stream>>>((const float4*)B, (const float4*)A, dinv,
                                           (const float4*)b1, (float4*)A, NH4); // A = h1

  // ----- GCN layer 2 -----
  k_gemm128  <<<ggrid, 256, 0, stream>>>(A, W2, B, n_tiles);
  k_fill4    <<<cb(NH4), 256, 0, stream>>>((float4*)A, NH4, 0.0f);
  k_scatter  <<<cb((long)E * 32), 256, 0, stream>>>(B, src, dst, dinv, A, E);
  k_finalize4<<<cb(NH4), 256, 0, stream>>>((const float4*)A, (const float4*)B, dinv,
                                           (const float4*)b2, (float4*)A, NH4); // A = h2

  // ----- global mean pool -----
  k_fill4    <<<cb((NG * HID + NG) / 4), 256, 0, stream>>>((float4*)pooled,
                                                           (NG * HID + NG) / 4, 0.0f);
  k_pool_sum4<<<cb(NH4), 256, 0, stream>>>((const float4*)A, bat, pooled, NH4);
  k_pool_cnt <<<cb(N), 256, 0, stream>>>(bat, cnt, N);
  k_pool_div4<<<cb(GH4), 256, 0, stream>>>((float4*)pooled, cnt, GH4);

  // ----- MLP head ----- (64 rows = 4 tiles)
  k_gemm128   <<<NG / 16, 256, 0, stream>>>(pooled, Wh1, ztmp, NG / 16);
  k_bias_relu4<<<cb(GH4), 256, 0, stream>>>((const float4*)ztmp, (const float4*)bh1,
                                            (float4*)z, GH4);
  k_head      <<<1, 64, 0, stream>>>(z, Wh2, bh2, out, NG);
}
